// CausalPerformer_30416958390632
// MI455X (gfx1250) — compile-verified, matrix operand order checked by reading, not run on hardware
//
#include <hip/hip_runtime.h>
#include <hip/hip_bf16.h>
#include <math.h>

// ---------------------------------------------------------------------------
// CausalPerformer on gfx1250 (MI455X):
//   qh/kh/vh = x @ W^T            -> WMMA bf16x3 GEMM (compute bound, 98.5% of FLOPs)
//   qp/kp    = norm(exp(-.5 (xh . omega^T)^2))   -> small VALU kernel
//   causal scan over 16x64 state  -> 3-pass chunked scan (chunk=64, 2048 WGs)
//   out      = attn @ w_o^T       -> same WMMA GEMM
// GEMM v2: 64x64 wave tile (48 WMMA per 32 ds_load_b128 per K-step),
//          register-pipelined global->LDS staging, XOR-swizzled bank-conflict-free LDS.
// ---------------------------------------------------------------------------

typedef __attribute__((ext_vector_type(16))) __bf16 v16bf;
typedef __attribute__((ext_vector_type(8)))  __bf16 v8bf;
typedef __attribute__((ext_vector_type(4)))  __bf16 v4bf;
typedef __attribute__((ext_vector_type(8)))  float  v8f;

#define BM 128
#define BN 256
#define BK 32

// Swizzled LDS element index for [row][elem] tiles of BK=32 bf16 per row (64B rows).
// 16B chunks are XOR-swizzled with (row>>2)&3 so a fragment read (16 lanes, 16B each,
// rows r..r+15, fixed chunk) covers all 64 LDS banks conflict-free. 16B alignment kept.
static __device__ __forceinline__ int swidx(int row, int elem) {
  const int chunk = elem >> 3;
  const int sub   = elem & 7;
  return row * BK + (((chunk ^ ((row >> 2) & 3)) << 3) | sub);
}

// Build a 16x32 bf16 WMMA operand fragment from a swizzled LDS tile.
// Per CDNA5 ISA 7.12.2 (16-bit operand, wave32): lanes 0-15 hold row=lane with
// K groups {0..7,16..23}; lanes 16-31 hold row=lane-16 with K {8..15,24..31}.
static __device__ __forceinline__ v16bf fragsw(const __bf16* base, int row, int kb) {
  const int sw = (row >> 2) & 3;
  const __bf16* rp = base + row * BK;
  v8bf a = *(const v8bf*)(rp + ((((kb >> 3)       ^ sw) << 3)));  // K = kb..kb+7
  v8bf b = *(const v8bf*)(rp + ((((kb + 16) >> 3) ^ sw) << 3));   // K = 16+kb..16+kb+7
  v16bf r;
#pragma unroll
  for (int e = 0; e < 8; ++e) { r[e] = a[e]; r[8 + e] = b[e]; }
  return r;
}

// C[m][n] = sum_k A[m][k] * W[n][k]   (A: MxK row-major, W: NxK row-major)
// bf16x3 split: a = hi+lo, b = hi+lo, accumulate hi*hi + hi*lo + lo*hi in fp32.
__global__ __launch_bounds__(256)
void gemm_abt(const float* __restrict__ A, const float* __restrict__ W,
              float* __restrict__ C, int M, int N, int K) {
  __shared__ __bf16 sAh[BM * BK];   //  8 KB
  __shared__ __bf16 sAl[BM * BK];   //  8 KB
  __shared__ __bf16 sBh[BN * BK];   // 16 KB
  __shared__ __bf16 sBl[BN * BK];   // 16 KB

  const int tid  = threadIdx.x;
  const int lane = tid & 31;
  const int wave = tid >> 5;
  const int wm   = wave >> 2;   // 0..1 : which 64-row slab of the 128-row block tile
  const int wn   = wave & 3;    // 0..3 : which 64-col slab of the 256-col block tile
  const int lr   = lane & 15;
  const int lh   = lane >> 4;

  const long m0 = (long)blockIdx.y * BM;
  const long n0 = (long)blockIdx.x * BN;

  v8f acc[4][4];
#pragma unroll
  for (int i = 0; i < 4; ++i)
#pragma unroll
    for (int j = 0; j < 4; ++j)
#pragma unroll
      for (int r = 0; r < 8; ++r) acc[i][j][r] = 0.0f;

  // Register-pipelined staging: tile kt+1's global loads issue before kt's WMMAs.
  float4 aReg[4];   // 128x32 A tile: 1024 float4 granules / 256 threads
  float4 wReg[8];   // 256x32 B tile: 2048 float4 granules / 256 threads

  const int ksteps = K / BK;

  // --- load tile 0 ---
#pragma unroll
  for (int i = 0; i < 4; ++i) {
    const int linear = tid + i * 256;
    aReg[i] = *(const float4*)&A[(m0 + (linear >> 3)) * K + ((linear & 7) << 2)];
  }
#pragma unroll
  for (int i = 0; i < 8; ++i) {
    const int linear = tid + i * 256;
    wReg[i] = *(const float4*)&W[(n0 + (linear >> 3)) * K + ((linear & 7) << 2)];
  }

  for (int kt = 0; kt < ksteps; ++kt) {
    // --- stage registers -> LDS as bf16 hi/lo (swizzled) ---
#pragma unroll
    for (int i = 0; i < 4; ++i) {
      const int linear = tid + i * 256;
      const int row = linear >> 3;
      const int cv  = (linear & 7) << 2;
      const float af[4] = {aReg[i].x, aReg[i].y, aReg[i].z, aReg[i].w};
      v4bf h4, l4;
#pragma unroll
      for (int e = 0; e < 4; ++e) {
        __bf16 h = (__bf16)af[e]; h4[e] = h; l4[e] = (__bf16)(af[e] - (float)h);
      }
      const int si = swidx(row, cv);
      *(v4bf*)&sAh[si] = h4;
      *(v4bf*)&sAl[si] = l4;
    }
#pragma unroll
    for (int i = 0; i < 8; ++i) {
      const int linear = tid + i * 256;
      const int row = linear >> 3;
      const int cv  = (linear & 7) << 2;
      const float wf[4] = {wReg[i].x, wReg[i].y, wReg[i].z, wReg[i].w};
      v4bf h4, l4;
#pragma unroll
      for (int e = 0; e < 4; ++e) {
        __bf16 h = (__bf16)wf[e]; h4[e] = h; l4[e] = (__bf16)(wf[e] - (float)h);
      }
      const int si = swidx(row, cv);
      *(v4bf*)&sBh[si] = h4;
      *(v4bf*)&sBl[si] = l4;
    }
    __syncthreads();

    // --- issue next tile's global loads; WMMA work below hides their latency ---
    if (kt + 1 < ksteps) {
      const long kofs = (long)(kt + 1) * BK;
#pragma unroll
      for (int i = 0; i < 4; ++i) {
        const int linear = tid + i * 256;
        aReg[i] = *(const float4*)&A[(m0 + (linear >> 3)) * K + kofs + ((linear & 7) << 2)];
      }
#pragma unroll
      for (int i = 0; i < 8; ++i) {
        const int linear = tid + i * 256;
        wReg[i] = *(const float4*)&W[(n0 + (linear >> 3)) * K + kofs + ((linear & 7) << 2)];
      }
    }

    // --- 48 WMMAs per K-step from 32 ds_load_b128 ---
    const int kb = lh * 8;
    v16bf bhf[4], blf[4];
#pragma unroll
    for (int ni = 0; ni < 4; ++ni) {
      const int r = wn * 64 + ni * 16 + lr;
      bhf[ni] = fragsw(sBh, r, kb);
      blf[ni] = fragsw(sBl, r, kb);
    }
#pragma unroll
    for (int mi = 0; mi < 4; ++mi) {
      const int r = wm * 64 + mi * 16 + lr;
      v16bf ahf = fragsw(sAh, r, kb);
      v16bf alf = fragsw(sAl, r, kb);
#pragma unroll
      for (int ni = 0; ni < 4; ++ni) {
        acc[mi][ni] = __builtin_amdgcn_wmma_f32_16x16x32_bf16(
            false, ahf, false, bhf[ni], (short)0, acc[mi][ni], false, false);
        acc[mi][ni] = __builtin_amdgcn_wmma_f32_16x16x32_bf16(
            false, ahf, false, blf[ni], (short)0, acc[mi][ni], false, false);
        acc[mi][ni] = __builtin_amdgcn_wmma_f32_16x16x32_bf16(
            false, alf, false, bhf[ni], (short)0, acc[mi][ni], false, false);
      }
    }
    __syncthreads();
  }

  // Epilogue: ISA C/D layout -> row = base + 8*(lane>=16) + vgpr, col = base + (lane&15)
#pragma unroll
  for (int mi = 0; mi < 4; ++mi) {
#pragma unroll
    for (int ni = 0; ni < 4; ++ni) {
      const long row0 = m0 + wm * 64 + mi * 16 + lh * 8;
      const long col  = n0 + wn * 64 + ni * 16 + lr;
#pragma unroll
      for (int r = 0; r < 8; ++r)
        C[(row0 + r) * N + col] = acc[mi][ni][r];
    }
  }
}

// qp/kp = normalize(exp(-0.5 * (xh . omega^T)^2)) over NBF=16 features.
__global__ __launch_bounds__(256)
void feat_kernel(const float* __restrict__ xh, const float* __restrict__ omega,
                 float* __restrict__ xp, int BLH) {
  __shared__ float som[16 * 64];
  for (int i = threadIdx.x; i < 1024; i += 256) som[i] = omega[i];
  __syncthreads();
  const int idx = blockIdx.x * 256 + threadIdx.x;
  if (idx >= BLH) return;
  const int  h  = idx & 15;          // H = 16
  const long bl = idx >> 4;
  const float* x = xh + bl * 1024 + h * 64;
  float f[16];
#pragma unroll
  for (int j = 0; j < 16; ++j) f[j] = 0.0f;
  for (int d = 0; d < 64; ++d) {
    const float xv = x[d];
#pragma unroll
    for (int j = 0; j < 16; ++j) f[j] += xv * som[j * 64 + d];
  }
  float s = 0.0f;
#pragma unroll
  for (int j = 0; j < 16; ++j) { f[j] = __expf(-0.5f * f[j] * f[j]); s += f[j]; }
  const float inv = 1.0f / (s + 1e-6f);
  float* o = xp + (long)idx * 16;
#pragma unroll
  for (int j = 0; j < 16; ++j) o[j] = f[j] * inv;
}

// Pass A: per-chunk totals of kp (16) and kp (x) vh (16x64). 64 threads = one d column each.
__global__ __launch_bounds__(64)
void scan_chunksum(const float* __restrict__ kp, const float* __restrict__ vh,
                   float* __restrict__ Ssum, float* __restrict__ zsum,
                   int L, int H, int NC, int CL) {
  const int cid = blockIdx.x;
  const int c   = cid % NC;
  const int bh  = cid / NC;
  const int h   = bh % H;
  const int b   = bh / H;
  const int d   = threadIdx.x;
  __shared__ float skp[16];
  float S[16];
#pragma unroll
  for (int f = 0; f < 16; ++f) S[f] = 0.0f;
  float zloc = 0.0f;
  const long l0 = (long)c * CL;
  for (int li = 0; li < CL; ++li) {
    const long l   = l0 + li;
    const long blh = ((long)b * L + l) * H + h;
    __syncthreads();
    if (d < 16) skp[d] = kp[blh * 16 + d];
    const float vd = vh[((long)b * L + l) * 1024 + h * 64 + d];
    __syncthreads();
#pragma unroll
    for (int f = 0; f < 16; ++f) S[f] += skp[f] * vd;
    if (d < 16) zloc += skp[d];
  }
  const long base = ((long)bh * NC + c) * 1024;
#pragma unroll
  for (int f = 0; f < 16; ++f) Ssum[base + f * 64 + d] = S[f];
  if (d < 16) zsum[((long)bh * NC + c) * 16 + d] = zloc;
}

// Pass B: in-place exclusive prefix over chunks (per b,h). Small and sequential.
__global__ __launch_bounds__(64)
void scan_prefix(float* __restrict__ Ssum, float* __restrict__ zsum, int NC) {
  const int bh = blockIdx.x;
  const int d  = threadIdx.x;
  float run[16];
#pragma unroll
  for (int f = 0; f < 16; ++f) run[f] = 0.0f;
  for (int c = 0; c < NC; ++c) {
    const long base = ((long)bh * NC + c) * 1024;
#pragma unroll
    for (int f = 0; f < 16; ++f) {
      const float t = Ssum[base + f * 64 + d];
      Ssum[base + f * 64 + d] = run[f];
      run[f] += t;
    }
  }
  if (d < 16) {
    float rz = 0.0f;
    for (int c = 0; c < NC; ++c) {
      const long zb = ((long)bh * NC + c) * 16 + d;
      const float t = zsum[zb];
      zsum[zb] = rz;
      rz += t;
    }
  }
}

// Pass C: inclusive scan within chunk from exclusive-prefixed start state; emit attn.
__global__ __launch_bounds__(64)
void scan_apply(const float* __restrict__ qp, const float* __restrict__ kp,
                const float* __restrict__ vh, const float* __restrict__ Ssum,
                const float* __restrict__ zsum, float* __restrict__ attn,
                int L, int H, int NC, int CL) {
  const int cid = blockIdx.x;
  const int c   = cid % NC;
  const int bh  = cid / NC;
  const int h   = bh % H;
  const int b   = bh / H;
  const int d   = threadIdx.x;
  __shared__ float skp[16], sqp[16], sz[16];
  float S[16];
  const long base = ((long)bh * NC + c) * 1024;
#pragma unroll
  for (int f = 0; f < 16; ++f) S[f] = Ssum[base + f * 64 + d];
  if (d < 16) sz[d] = zsum[((long)bh * NC + c) * 16 + d];
  const long l0 = (long)c * CL;
  for (int li = 0; li < CL; ++li) {
    const long l   = l0 + li;
    const long blh = ((long)b * L + l) * H + h;
    __syncthreads();                       // protect previous iteration's reads
    if (d < 16) { skp[d] = kp[blh * 16 + d]; sqp[d] = qp[blh * 16 + d]; }
    const float vd = vh[((long)b * L + l) * 1024 + h * 64 + d];
    __syncthreads();
    if (d < 16) sz[d] += skp[d];           // inclusive z before denominator
    __syncthreads();
    float num = 0.0f, den = 0.0f;
#pragma unroll
    for (int f = 0; f < 16; ++f) {
      S[f] += skp[f] * vd;                 // inclusive state before numerator
      num  += sqp[f] * S[f];
      den  += sqp[f] * sz[f];
    }
    attn[((long)b * L + l) * 1024 + h * 64 + d] = num / (den + 1e-6f);
  }
}

extern "C" void kernel_launch(void* const* d_in, const int* in_sizes, int n_in,
                              void* d_out, int out_size, void* d_ws, size_t ws_size,
                              hipStream_t stream) {
  (void)in_sizes; (void)n_in; (void)out_size; (void)ws_size;
  const float* q     = (const float*)d_in[0];
  const float* k     = (const float*)d_in[1];
  const float* v     = (const float*)d_in[2];
  const float* w_q   = (const float*)d_in[3];
  const float* w_k   = (const float*)d_in[4];
  const float* w_v   = (const float*)d_in[5];
  const float* w_o   = (const float*)d_in[6];
  const float* omega = (const float*)d_in[7];
  float* out = (float*)d_out;

  const int B = 2, L = 4096, D = 1024, H = 16;
  const int CL = 64, NC = L / CL;       // chunked scan
  const long MR = (long)B * L;          // 8192 GEMM rows

  // Workspace layout (floats). Total ~120 MB. attn output reuses qh (dead after feat).
  float* ws   = (float*)d_ws;
  float* qh   = ws;                 //  8,388,608
  float* kh   = ws + 8388608L;      //  8,388,608
  float* vh   = ws + 16777216L;     //  8,388,608
  float* qp   = ws + 25165824L;     //  2,097,152
  float* kp   = ws + 27262976L;     //  2,097,152
  float* Ssum = ws + 29360128L;     //  2,097,152
  float* zsum = ws + 31457280L;     //     32,768
  float* attn = qh;

  const dim3 ggrid(D / BN, (int)(MR / BM));   // (4, 64)
  const dim3 gblk(256);

  // 1) projections
  gemm_abt<<<ggrid, gblk, 0, stream>>>(q, w_q, qh, (int)MR, D, D);
  gemm_abt<<<ggrid, gblk, 0, stream>>>(k, w_k, kh, (int)MR, D, D);
  gemm_abt<<<ggrid, gblk, 0, stream>>>(v, w_v, vh, (int)MR, D, D);

  // 2) random-feature maps
  const int BLH = (int)(MR * H);              // 131072
  feat_kernel<<<BLH / 256, 256, 0, stream>>>(qh, omega, qp, BLH);
  feat_kernel<<<BLH / 256, 256, 0, stream>>>(kh, omega, kp, BLH);

  // 3) chunked causal scan
  scan_chunksum<<<B * H * NC, 64, 0, stream>>>(kp, vh, Ssum, zsum, L, H, NC, CL);
  scan_prefix<<<B * H, 64, 0, stream>>>(Ssum, zsum, NC);
  scan_apply<<<B * H * NC, 64, 0, stream>>>(qp, kp, vh, Ssum, zsum, attn, L, H, NC, CL);

  // 4) output projection
  gemm_abt<<<ggrid, gblk, 0, stream>>>(attn, w_o, out, (int)MR, D, D);
}
